// SelfAttention_17935783428566
// MI455X (gfx1250) — matrix operand order repaired and checked by reading, not compiled, not run on hardware
//
#include <hip/hip_runtime.h>
#include <hip/hip_bf16.h>
#include <math.h>

// ---------------------------------------------------------------------------
// Types for CDNA5 WMMA (wave32): v_wmma_f32_16x16x32_bf16
// ---------------------------------------------------------------------------
typedef __bf16 bf16_t;
typedef bf16_t v16bf __attribute__((ext_vector_type(16)));
typedef float  v8f   __attribute__((ext_vector_type(8)));

#define WMMA_BF16(A, Bm, C) \
  __builtin_amdgcn_wmma_f32_16x16x32_bf16(false, (A), false, (Bm), (short)0, (C), false, false)

// Problem constants (from the reference)
constexpr int   kB   = 2;
constexpr int   kS   = 2048;
constexpr int   kD   = 1024;
constexpr int   kH   = 16;
constexpr int   kDh  = 64;
constexpr float kSmScale = 0.125f;  // 1/sqrt(64)

static __device__ __forceinline__ int lane_id() { return (int)(threadIdx.x & 31u); }

// ---------------------------------------------------------------------------
// CDNA5 async copy: global -> LDS, tracked by ASYNCcnt.
// LDS operand is the raw LDS byte offset; for a generic pointer to __shared__
// the flat aperture lives entirely in addr[63:32], so addr[31:0] IS the offset.
// ---------------------------------------------------------------------------
static __device__ __forceinline__ unsigned lds_off_of(const void* p) {
  return (unsigned)(uintptr_t)p;
}
static __device__ __forceinline__ void async_load_b128(unsigned lds_byte_off,
                                                       const void* gptr) {
  asm volatile("global_load_async_to_lds_b128 %0, %1, off"
               :: "v"(lds_byte_off), "v"((unsigned long long)(uintptr_t)gptr)
               : "memory");
}
static __device__ __forceinline__ void wait_async0() {
  asm volatile("s_wait_asynccnt 0" ::: "memory");
}

// ---------------------------------------------------------------------------
// Fragment loaders implementing the CDNA5 VGPR layouts (ISA 7.12.2):
// A 16x32 (16-bit): lane L holds row M=L&15; half h=L>>4:
//   elem pair 2v,2v+1 -> K = ((v&4)<<2) + 8*h + 2*(v&3) + (elem&1)
// B 32x16 (16-bit): lane L holds col N=L&15; elem e -> K = 8*(L>>4) + e + (e>=8 ? 8 : 0)
// ---------------------------------------------------------------------------
template <typename T>
static __device__ __forceinline__ v16bf load_a_rowmajor(const T* __restrict__ src,
                                                        int ld, int row0, int k0) {
  const int lane = lane_id();
  const T* p = src + (size_t)(row0 + (lane & 15)) * ld + k0 + ((lane >> 4) & 1) * 8;
  v16bf f;
#pragma unroll
  for (int vg = 0; vg < 8; ++vg) {
    const int k = ((vg & 4) << 2) + ((vg & 3) << 1);  // 0,2,4,6,16,18,20,22
    f[2 * vg]     = (bf16_t)p[k];
    f[2 * vg + 1] = (bf16_t)p[k + 1];
  }
  return f;
}

template <typename T>
static __device__ __forceinline__ v16bf load_bT_rowmajor(const T* __restrict__ src,
                                                         int ld, int col0, int k0) {
  const int lane = lane_id();
  const T* p = src + (size_t)(col0 + (lane & 15)) * ld + k0 + ((lane >> 4) & 1) * 8;
  v16bf f;
#pragma unroll
  for (int e = 0; e < 16; ++e) f[e] = (bf16_t)p[e < 8 ? e : e + 8];
  return f;
}

// ---------------------------------------------------------------------------
// Elementwise fp32 -> bf16 conversion (done once; removes cvt from GEMM loops)
// ---------------------------------------------------------------------------
__global__ __launch_bounds__(256) void cvt_f32_bf16_kernel(
    const float* __restrict__ in, bf16_t* __restrict__ outp, int n) {
  const int i = (int)(blockIdx.x * blockDim.x + threadIdx.x) * 4;
  if (i + 3 < n) {
    const float4 v = *(const float4*)(in + i);
    outp[i]     = (bf16_t)v.x;
    outp[i + 1] = (bf16_t)v.y;
    outp[i + 2] = (bf16_t)v.z;
    outp[i + 3] = (bf16_t)v.w;
  }
}

// ---------------------------------------------------------------------------
// Shared GEMM core: block = 4 waves computing a 64(M) x 64(N) tile.
// The 64x32 bf16 weight tile (B operand, rows = output cols) is staged into
// LDS with async b128 copies, double buffered; each wave computes 16x64
// with 4 independent accumulators (4 back-to-back WMMAs per k-step).
// ---------------------------------------------------------------------------
static __device__ __forceinline__ void gemm_wave_16x64(
    const bf16_t* __restrict__ A, int lda, int m0,   // per-wave A rows (bf16)
    const bf16_t* __restrict__ W, int ldw, int n0,   // shared B rows   (bf16)
    bf16_t (*ldsW)[64][32],                          // [2][64][32] ping-pong
    int K, v8f acc[4]) {
  const int tid = (int)threadIdx.x;

  auto issue = [&](int buf, int k0) {
    // 64 rows x 32 bf16 = 256 x 16B chunks, 128 threads x 2 chunks
#pragma unroll
    for (int i = 0; i < 2; ++i) {
      const int c   = tid + i * 128;
      const int row = c >> 2, seg = c & 3;
      const bf16_t* g = W + (size_t)(n0 + row) * ldw + k0 + seg * 8;
      async_load_b128(lds_off_of(&ldsW[buf][row][seg * 8]), g);
    }
  };

  issue(0, 0);
  for (int k0 = 0, it = 0; k0 < K; k0 += 32, ++it) {
    const int cur = it & 1;
    wait_async0();       // this wave's async copies done
    __syncthreads();     // whole tile visible to all 4 waves
    if (k0 + 32 < K) issue(cur ^ 1, k0 + 32);

    const v16bf a = load_a_rowmajor(A, lda, m0, k0);
#pragma unroll
    for (int t = 0; t < 4; ++t) {
      v16bf b = load_bT_rowmajor(&ldsW[cur][0][0], 32, t * 16, 0);
      acc[t] = WMMA_BF16(a, b, acc[t]);
    }
    __syncthreads();     // all reads of `cur` done before it is refilled
  }
}

// ---------------------------------------------------------------------------
// Kernel 1: QKV projection. qkv = x @ in_proj_w^T + b  (4096 x 3072)
// Scatter to Q [B*H,S,64], K [B*H,S,64], V^T [B*H,64,S] (all bf16).
// ---------------------------------------------------------------------------
__global__ __launch_bounds__(128) void qkv_proj_kernel(
    const bf16_t* __restrict__ xbf,   // [4096, 1024] bf16
    const bf16_t* __restrict__ wbf,   // [3072, 1024] bf16
    const float*  __restrict__ bias,  // [3072]
    bf16_t* __restrict__ qbuf,
    bf16_t* __restrict__ kbuf,
    bf16_t* __restrict__ vtbuf) {
  __shared__ bf16_t ldsW[2][64][32];
  const int wid = (int)(threadIdx.x >> 5);
  const int n0  = (int)blockIdx.x * 64;            // 48 tiles -> 3072
  const int m0  = (int)blockIdx.y * 64 + wid * 16; // 64 tiles -> 4096

  v8f acc[4] = {};
  gemm_wave_16x64(xbf, kD, m0, wbf, kD, n0, ldsW, kD, acc);

  const int lane = lane_id();
  const int n = lane & 15, h = (lane >> 4) & 1;
  const int sec  = n0 >> 10;        // constant across the 64-col tile
  const int head = (n0 >> 6) & 15;  // 64-aligned -> constant
#pragma unroll
  for (int t = 0; t < 4; ++t) {
    const int dh = t * 16 + n;
    const float bn = bias[n0 + t * 16 + n];
#pragma unroll
    for (int r = 0; r < 8; ++r) {
      const int m  = m0 + r + h * 8;
      const int bb = m >> 11;
      const int s  = m & 2047;
      const bf16_t v = (bf16_t)(acc[t][r] + bn);
      const size_t bh = (size_t)(bb * kH + head);
      if (sec == 0)      qbuf[(bh * kS + s) * kDh + dh] = v;
      else if (sec == 1) kbuf[(bh * kS + s) * kDh + dh] = v;
      else               vtbuf[(bh * kDh + dh) * kS + s] = v;
    }
  }
}

// ---------------------------------------------------------------------------
// Kernel 2: causal flash attention. One wave per (b, head, 16-query tile).
// Key blocks of 32 -> 4 WMMAs for scores, online softmax, P staged via LDS
// (C-layout -> A-layout), 4 WMMAs for P@V against contiguous V^T rows.
// ---------------------------------------------------------------------------
__global__ __launch_bounds__(128) void attn_kernel(
    const bf16_t* __restrict__ qbuf,   // [B*H, S, 64]
    const bf16_t* __restrict__ kbuf,   // [B*H, S, 64]
    const bf16_t* __restrict__ vtbuf,  // [B*H, 64, S]
    bf16_t* __restrict__ ctx) {        // [B*S, 1024]
  __shared__ bf16_t pLds[4][16][32];

  const int wid  = (int)(threadIdx.x >> 5);
  const int lane = lane_id();
  const int tq   = (int)blockIdx.x * 4 + wid;  // 0..127
  const int head = (int)blockIdx.y;
  const int b    = (int)blockIdx.z;
  const size_t bh = (size_t)(b * kH + head);
  const bf16_t* Q  = qbuf  + bh * kS * kDh;
  const bf16_t* K  = kbuf  + bh * kS * kDh;
  const bf16_t* Vt = vtbuf + bh * kDh * kS;
  const int q0 = tq * 16;

  const v16bf qa0 = load_a_rowmajor(Q, kDh, q0, 0);
  const v16bf qa1 = load_a_rowmajor(Q, kDh, q0, 32);

  v8f o[4] = {};
  float mrow[8], lrow[8];
#pragma unroll
  for (int r = 0; r < 8; ++r) { mrow[r] = -__builtin_inff(); lrow[r] = 0.f; }

  const int h = (lane >> 4) & 1;
  const int n = lane & 15;
  const int kend = q0 + 16;  // causal bound (exclusive)

  for (int k0 = 0; k0 < kend; k0 += 32) {
    // prefetch the next key block while we compute this one
    if (k0 + 32 < kend) {
      __builtin_prefetch(K + (size_t)(k0 + 32 + (lane & 31)) * kDh, 0, 1);
      __builtin_prefetch(Vt + (size_t)(lane & 31) * kS + k0 + 32, 0, 1);
    }

    // ---- scores for 32 keys: two 16x16 tiles ----
    v8f s0 = {}, s1 = {};
    {
      v16bf kb0 = load_bT_rowmajor(K, kDh, k0, 0);
      v16bf kb1 = load_bT_rowmajor(K, kDh, k0, 32);
      s0 = WMMA_BF16(qa0, kb0, s0);
      s0 = WMMA_BF16(qa1, kb1, s0);
      v16bf kb2 = load_bT_rowmajor(K, kDh, k0 + 16, 0);
      v16bf kb3 = load_bT_rowmajor(K, kDh, k0 + 16, 32);
      s1 = WMMA_BF16(qa0, kb2, s1);
      s1 = WMMA_BF16(qa1, kb3, s1);
    }

    // ---- scale + causal mask + online softmax update ----
#pragma unroll
    for (int r = 0; r < 8; ++r) {
      const int qi = q0 + r + h * 8;
      const float v0 = (k0 + n      <= qi) ? s0[r] * kSmScale : -__builtin_inff();
      const float v1 = (k0 + 16 + n <= qi) ? s1[r] * kSmScale : -__builtin_inff();

      float tmax = fmaxf(v0, v1);
#pragma unroll
      for (int xm = 8; xm >= 1; xm >>= 1) tmax = fmaxf(tmax, __shfl_xor(tmax, xm, 32));
      const float mnew  = fmaxf(mrow[r], tmax);
      const float alpha = __expf(mrow[r] - mnew);
      const float p0 = __expf(v0 - mnew);
      const float p1 = __expf(v1 - mnew);
      float psum = p0 + p1;
#pragma unroll
      for (int xm = 8; xm >= 1; xm >>= 1) psum += __shfl_xor(psum, xm, 32);
      lrow[r] = lrow[r] * alpha + psum;
      mrow[r] = mnew;
#pragma unroll
      for (int t = 0; t < 4; ++t) o[t][r] *= alpha;

      pLds[wid][r + h * 8][n]      = (bf16_t)p0;
      pLds[wid][r + h * 8][16 + n] = (bf16_t)p1;
    }
    __builtin_amdgcn_wave_barrier();  // same-wave DS ops are in-order

    // ---- O += P @ V ----
    {
      const v16bf pa = load_a_rowmajor(&pLds[wid][0][0], 32, 0, 0);
#pragma unroll
      for (int t = 0; t < 4; ++t) {
        v16bf vb = load_bT_rowmajor(Vt, kS, t * 16, k0);
        o[t] = WMMA_BF16(pa, vb, o[t]);
      }
    }
    __builtin_amdgcn_wave_barrier();
  }

  // ---- normalize and write context (bf16 for the out-proj GEMM) ----
#pragma unroll
  for (int r = 0; r < 8; ++r) {
    const float inv = 1.0f / lrow[r];
    const int s = q0 + r + h * 8;
    const size_t row = (size_t)(b * kS + s) * kD + head * kDh;
#pragma unroll
    for (int t = 0; t < 4; ++t)
      ctx[row + t * 16 + n] = (bf16_t)(o[t][r] * inv);
  }
}

// ---------------------------------------------------------------------------
// Kernel 3: output projection. out = ctx @ out_proj_w^T + b  (fp32 output)
// ---------------------------------------------------------------------------
__global__ __launch_bounds__(128) void out_proj_kernel(
    const bf16_t* __restrict__ ctx,  // [4096, 1024] bf16
    const bf16_t* __restrict__ wbf,  // [1024, 1024] bf16
    const float* __restrict__ bias,  // [1024]
    float* __restrict__ out) {       // [4096, 1024] fp32
  __shared__ bf16_t ldsW[2][64][32];
  const int wid = (int)(threadIdx.x >> 5);
  const int n0  = (int)blockIdx.x * 64;            // 16 tiles -> 1024
  const int m0  = (int)blockIdx.y * 64 + wid * 16; // 64 tiles -> 4096

  v8f acc[4] = {};
  gemm_wave_16x64(ctx, kD, m0, wbf, kD, n0, ldsW, kD, acc);

  const int lane = lane_id();
  const int n = lane & 15, h = (lane >> 4) & 1;
#pragma unroll
  for (int t = 0; t < 4; ++t) {
    const float bn = bias[n0 + t * 16 + n];
#pragma unroll
    for (int r = 0; r < 8; ++r)
      out[(size_t)(m0 + r + h * 8) * kD + n0 + t * 16 + n] = acc[t][r] + bn;
  }
}

// ---------------------------------------------------------------------------
// Launch
// ---------------------------------------------------------------------------
extern "C" void kernel_launch(void* const* d_in, const int* in_sizes, int n_in,
                              void* d_out, int out_size, void* d_ws, size_t ws_size,
                              hipStream_t stream) {
  (void)in_sizes; (void)n_in; (void)out_size; (void)ws_size;
  const float* x     = (const float*)d_in[0];
  const float* in_w  = (const float*)d_in[1];
  const float* in_b  = (const float*)d_in[2];
  const float* out_w = (const float*)d_in[3];
  const float* out_b = (const float*)d_in[4];
  float* out = (float*)d_out;

  // Workspace layout (bf16 elements):
  //   Q | K | V^T | ctx | x_bf16 | in_w_bf16 | out_w_bf16   (~50 MB total)
  bf16_t* ws = (bf16_t*)d_ws;
  const size_t per = (size_t)kB * kH * kS * kDh;  // 4,194,304
  bf16_t* qbuf  = ws;
  bf16_t* kbuf  = ws + per;
  bf16_t* vtbuf = ws + 2 * per;
  bf16_t* ctx   = ws + 3 * per;
  bf16_t* xbf   = ws + 4 * per;                   // 4096*1024
  bf16_t* winb  = xbf + (size_t)4096 * 1024;      // 3072*1024
  bf16_t* woutb = winb + (size_t)3072 * 1024;     // 1024*1024

  // 0) one-time fp32 -> bf16 conversion of activations and weights
  cvt_f32_bf16_kernel<<<4096, 256, 0, stream>>>(x, xbf, 4096 * 1024);
  cvt_f32_bf16_kernel<<<3072, 256, 0, stream>>>(in_w, winb, 3072 * 1024);
  cvt_f32_bf16_kernel<<<1024, 256, 0, stream>>>(out_w, woutb, 1024 * 1024);

  // 1) QKV projection: 48 x 64 block tiles (64x64 each), async-LDS staged B
  qkv_proj_kernel<<<dim3(48, 64), 128, 0, stream>>>(xbf, winb, in_b, qbuf, kbuf, vtbuf);
  // 2) causal flash attention
  attn_kernel<<<dim3(32, 16, 2), 128, 0, stream>>>(qbuf, kbuf, vtbuf, ctx);
  // 3) output projection: 16 x 64 block tiles
  out_proj_kernel<<<dim3(16, 64), 128, 0, stream>>>(ctx, woutb, out_b, out);
}
